// KNNFeatureBlock_82729660055793
// MI455X (gfx1250) — matrix operand order repaired and checked by reading, not compiled, not run on hardware
//
#include <hip/hip_runtime.h>

// ---------------------------------------------------------------------------
// KNN feature block for MI455X (gfx1250, wave32, WMMA).
//   1. convert_w    : w2/w3/w4 f32 -> bf16 in ws (L2-resident, ~0.7MB)
//   2. knn_kernel   : per-query bitonic top-128, gather centered coords,
//                     per-group 3x3 moment partials (for analytic BN1)
//   3. reduce_bn1   : fixed-order sum of 4096 moment partials
//   4. bn1_finalize : analytic BN1 scale/shift (BN of a linear map of x)
//   5. stage<0>     : conv1+bn1+relu -> WMMA conv2 -> max/concat ->
//                     WMMA conv3 -> per-channel sum/sumsq partials (BN2)
//   6. reduce_bn2   : fixed-order sum over groups (deterministic)
//   7. bn2_finalize : BN2 scale/shift
//   8. stage<1>     : recompute conv1..conv3, bn2+relu -> WMMA conv4 ->
//                     max over k -> feat (B,C,N)
// ~380 GFLOP of GEMM through v_wmma_f32_16x16x32_bf16 (f32 accum); N-loop
// blocked 2-wide so each A-fragment (ds_load_b128 pair) feeds two
// independent accumulator chains.
// ---------------------------------------------------------------------------

typedef __attribute__((ext_vector_type(16))) __bf16 v16bf;
typedef __attribute__((ext_vector_type(8)))  float  v8f;

#define NPTS   2048
#define KNB    128
#define NBATCH 2
#define NGRP   (NBATCH * NPTS)   // 4096
#define CNT_BN 524288.0f         // NGRP * KNB

// ---- workspace byte offsets ------------------------------------------------
#define WS_GP   0                       // grouped centered pos f32: 4096*128*3
#define WS_W2   6291456                 // 256*128 bf16
#define WS_W3   6356992                 // 512*512 bf16
#define WS_W4   6881280                 // 128*512 bf16
#define WS_F    7012352                 // float region base
// float-region indices (in floats, from WS_F)
#define FI_SC1     0      // 128
#define FI_SH1     128    // 128
#define FI_SC2     256    // 512
#define FI_SH2     768    // 512
#define FI_STAT9   1280   // 9
#define FI_BN2ACC  1296   // 1024
#define FI_BN1PART 4096   // 4096*9
#define FI_BN2PART 49152  // 4096*1024

// ---- dynamic LDS pool byte offsets (stage kernel) --------------------------
#define OFF_A1    0        // 128x128 bf16  (32768)
#define OFF_Z     32768    // 128x512 bf16  (131072)
#define OFF_SCR   163840   // fgtmp[16][256] f32 / ftmp[16][128] f32 (16384)
#define OFF_SMALL 180224   // fg/xloc/w1/sc/sh (12288)
#define OFF_A3    192512   // stage C: 128x512 bf16; stage B: stat stage 64KB
#define SMEM_B    258048
#define SMEM_C    323584

union FragU { uint4 u[2]; v16bf v; };

__device__ __forceinline__ v16bf frag_a_lds(const __bf16* base, int row0,
                                            int kb, int stride) {
  const int lane = threadIdx.x & 31;
  const int m = lane & 15, h = lane >> 4;
  // ISA 16-bit A 16x32 layout: lanes 0-15 hold K {kb..kb+7, kb+16..kb+23},
  // lanes 16-31 hold K {kb+8..kb+15, kb+24..kb+31}; row = row0 + (lane&15).
  const __bf16* p = base + (size_t)(row0 + m) * stride + kb + h * 8;
  FragU f;
  f.u[0] = *(const uint4*)p;          // ds_load_b128
  f.u[1] = *(const uint4*)(p + 16);
  return f.v;
}

__device__ __forceinline__ v16bf frag_b_glb(const __bf16* w, int n0,
                                            int kb, int stride) {
  const int lane = threadIdx.x & 31;
  const int m = lane & 15, h = lane >> 4;
  // B (KxN): lane owns column n0+m; lanes 0-15 K {kb..kb+7, kb+16..kb+23}, 16-31 K {kb+8..kb+15, kb+24..kb+31}.
  const __bf16* p = w + (size_t)(n0 + m) * stride + kb + h * 8;
  FragU f;
  f.u[0] = *(const uint4*)p;          // global_load_b128
  f.u[1] = *(const uint4*)(p + 16);
  return f.v;
}

__device__ __forceinline__ v8f wmma_bf16(v16bf a, v16bf b, v8f c) {
  return __builtin_amdgcn_wmma_f32_16x16x32_bf16(
      false, a, false, b, (short)0, c, false, false);
}

// ---------------------------------------------------------------------------
__global__ __launch_bounds__(256) void convert_w(
    const float* __restrict__ w2, const float* __restrict__ w3,
    const float* __restrict__ w4, __bf16* o2, __bf16* o3, __bf16* o4) {
  const int n2 = 256 * 128, n3 = 512 * 512, n4 = 128 * 512;
  int i = blockIdx.x * 256 + threadIdx.x;
  if (i < n2) o2[i] = (__bf16)w2[i];
  int j = i - n2;
  if (j >= 0 && j < n3) o3[j] = (__bf16)w3[j];
  int l = i - n2 - n3;
  if (l >= 0 && l < n4) o4[l] = (__bf16)w4[l];
}

// ---------------------------------------------------------------------------
// One workgroup per query point: pos[b] in LDS, bitonic sort of 2048
// (d2, idx) pairs ascending with index tie-break (matches lax.top_k order).
__global__ __launch_bounds__(256) void knn_kernel(
    const float* __restrict__ pos, float* __restrict__ gp,
    int* __restrict__ outidx, float* __restrict__ bn1part) {
  __shared__ float ps[NPTS * 3];
  __shared__ float key[NPTS];
  __shared__ int   val[NPTS];
  __shared__ float mred[3 * KNB];

  const int g = blockIdx.x, b = g >> 11, n = g & (NPTS - 1);
  const int tid = threadIdx.x;
  const float* pb = pos + (size_t)b * NPTS * 3;
  for (int i = tid; i < NPTS * 3; i += 256) ps[i] = pb[i];
  __syncthreads();
  const float qx = ps[n * 3], qy = ps[n * 3 + 1], qz = ps[n * 3 + 2];
  for (int i = tid; i < NPTS; i += 256) {
    float dx = ps[i * 3] - qx, dy = ps[i * 3 + 1] - qy, dz = ps[i * 3 + 2] - qz;
    key[i] = dx * dx + dy * dy + dz * dz;
    val[i] = i;
  }
  for (int sz = 2; sz <= NPTS; sz <<= 1) {
    for (int st = sz >> 1; st > 0; st >>= 1) {
      __syncthreads();
      for (int i = tid; i < NPTS; i += 256) {
        int j = i ^ st;
        if (j > i) {
          bool up = ((i & sz) == 0);
          float ki = key[i], kj = key[j];
          int vi = val[i], vj = val[j];
          bool gt = (ki > kj) || (ki == kj && vi > vj);
          if (gt == up) { key[i] = kj; key[j] = ki; val[i] = vj; val[j] = vi; }
        }
      }
    }
  }
  __syncthreads();
  if (tid < KNB) {
    int id = val[tid];
    outidx[(size_t)g * KNB + tid] = id;
    float dx = ps[id * 3] - qx, dy = ps[id * 3 + 1] - qy, dz = ps[id * 3 + 2] - qz;
    size_t o = (size_t)g * KNB * 3 + tid * 3;
    gp[o] = dx; gp[o + 1] = dy; gp[o + 2] = dz;
    mred[tid] = dx; mred[KNB + tid] = dy; mred[2 * KNB + tid] = dz;
  }
  __syncthreads();
  if (tid == 0) {  // deterministic fixed-order partial moments
    float sx = 0, sy = 0, sz = 0, sxx = 0, syy = 0, szz = 0,
          sxy = 0, sxz = 0, syz = 0;
    for (int j = 0; j < KNB; ++j) {
      float x = mred[j], y = mred[KNB + j], z = mred[2 * KNB + j];
      sx += x; sy += y; sz += z;
      sxx += x * x; syy += y * y; szz += z * z;
      sxy += x * y; sxz += x * z; syz += y * z;
    }
    float* p = bn1part + (size_t)g * 9;
    p[0] = sx; p[1] = sy; p[2] = sz; p[3] = sxx; p[4] = syy; p[5] = szz;
    p[6] = sxy; p[7] = sxz; p[8] = syz;
  }
}

__global__ void reduce_bn1(const float* __restrict__ part, float* __restrict__ s9) {
  int t = threadIdx.x;
  if (t < 9) {
    float s = 0.f;
    for (int g = 0; g < NGRP; ++g) s += part[(size_t)g * 9 + t];
    s9[t] = s;
  }
}

// BN1 over h1 = w1.x + b1: E[h] and E[h^2] are linear/quadratic in the
// global 3x3 moment matrix of centered grouped coords.
__global__ void bn1_finalize(const float* __restrict__ s9,
                             const float* __restrict__ w1,
                             const float* __restrict__ b1,
                             const float* __restrict__ g1,
                             const float* __restrict__ be1,
                             float* __restrict__ sc1, float* __restrict__ sh1) {
  int c = threadIdx.x;
  if (c >= 128) return;
  float mx = s9[0] / CNT_BN, my = s9[1] / CNT_BN, mz = s9[2] / CNT_BN;
  float Mxx = s9[3] / CNT_BN, Myy = s9[4] / CNT_BN, Mzz = s9[5] / CNT_BN;
  float Mxy = s9[6] / CNT_BN, Mxz = s9[7] / CNT_BN, Myz = s9[8] / CNT_BN;
  float w0 = w1[c * 3], w1_ = w1[c * 3 + 1], w2_ = w1[c * 3 + 2], bb = b1[c];
  float wm = w0 * mx + w1_ * my + w2_ * mz;
  float mean = wm + bb;
  float e2 = w0 * w0 * Mxx + w1_ * w1_ * Myy + w2_ * w2_ * Mzz +
             2.f * (w0 * w1_ * Mxy + w0 * w2_ * Mxz + w1_ * w2_ * Myz) +
             2.f * bb * wm + bb * bb;
  float var = e2 - mean * mean;
  float sc = g1[c] * rsqrtf(var + 1e-5f);
  sc1[c] = sc;
  sh1[c] = sc * (bb - mean) + be1[c];  // a1 = sc*(w.x) + sh  (b1 folded)
}

// ---------------------------------------------------------------------------
// Main per-group pipeline. One workgroup (8 waves) per group; each wave owns
// a 16-row (k-point) tile and walks N in 2-tile blocks (shared A-fragment,
// two independent WMMA accumulator chains). WITH_OUT=0: BN2 partials.
// WITH_OUT=1: apply BN2+relu, conv4, column max -> feat.
template <int WITH_OUT>
__global__ __launch_bounds__(256) void stage_kernel(
    const float* __restrict__ gp, const float* __restrict__ w1,
    const __bf16* __restrict__ w2bf, const __bf16* __restrict__ w3bf,
    const __bf16* __restrict__ w4bf,
    const float* __restrict__ sc1g, const float* __restrict__ sh1g,
    const float* __restrict__ b2, const float* __restrict__ b3,
    const float* __restrict__ sc2g, const float* __restrict__ sh2g,
    const float* __restrict__ b4,
    float* __restrict__ bn2part, float* __restrict__ outfeat) {
  extern __shared__ char smem[];
  __bf16* a1  = (__bf16*)(smem + OFF_A1);    // [128][128]
  __bf16* zb  = (__bf16*)(smem + OFF_Z);     // [128][512] (fg | h2)
  float*  scr = (float*)(smem + OFF_SCR);    // fgtmp[16][256] / ftmp[16][128]
  float*  fg    = (float*)(smem + OFF_SMALL);
  float*  xloc  = fg + 256;                  // 384
  float*  w1loc = xloc + 384;                // 384
  float*  sc1l  = w1loc + 384;               // 128
  float*  sh1l  = sc1l + 128;                // 128
  float*  sc2l  = sh1l + 128;                // 512
  float*  sh2l  = sc2l + 512;                // 512
  __bf16* a3    = (__bf16*)(smem + OFF_A3);  // stage C [128][512]
  float*  statS = (float*)(smem + OFF_A3);   // stage B [16][512]
  float*  statSS = statS + 16 * 512;

  const int g = blockIdx.x, tid = threadIdx.x;
  const int wv = tid >> 5, lane = tid & 31, m = lane & 15, h = lane >> 4;
  const int mr = wv * 16;

  // ---- preload per-group coords + small params ----
  for (int i = tid; i < 384; i += 256) {
    xloc[i] = gp[(size_t)g * 384 + i];
    w1loc[i] = w1[i];
  }
  for (int i = tid; i < 128; i += 256) { sc1l[i] = sc1g[i]; sh1l[i] = sh1g[i]; }
  if (WITH_OUT) {
    for (int i = tid; i < 512; i += 256) { sc2l[i] = sc2g[i]; sh2l[i] = sh2g[i]; }
  }
  __syncthreads();

  // ---- conv1 (K=3, VALU) + BN1 + relu -> a1 bf16 [k=128][c=128] ----
  for (int e = tid; e < 128 * 128; e += 256) {
    int r = e >> 7, c = e & 127;
    float v = xloc[r * 3] * w1loc[c * 3] + xloc[r * 3 + 1] * w1loc[c * 3 + 1] +
              xloc[r * 3 + 2] * w1loc[c * 3 + 2];
    v = fmaxf(v * sc1l[c] + sh1l[c], 0.f);
    a1[e] = (__bf16)v;
  }
  __syncthreads();

  // ---- conv2: M=128 N=256 K=128 (WMMA) -> h2 into zb cols 256.., fg max ----
  float* fgtmp = scr;  // [16][256]
  for (int nt = 0; nt < 16; nt += 2) {
    v8f acc0 = {0.f, 0.f, 0.f, 0.f, 0.f, 0.f, 0.f, 0.f};
    v8f acc1 = acc0;
#pragma unroll
    for (int ks = 0; ks < 4; ++ks) {
      v16bf a = frag_a_lds(a1, mr, ks * 32, 128);
      acc0 = wmma_bf16(a, frag_b_glb(w2bf, nt * 16, ks * 32, 128), acc0);
      acc1 = wmma_bf16(a, frag_b_glb(w2bf, (nt + 1) * 16, ks * 32, 128), acc1);
    }
#pragma unroll
    for (int t = 0; t < 2; ++t) {
      v8f acc = t ? acc1 : acc0;
      int col = (nt + t) * 16 + m;
      float bias = b2[col];
      float mx = -3.4e38f;
#pragma unroll
      for (int i = 0; i < 8; ++i) {
        float v = acc[i] + bias;
        zb[(size_t)(mr + h * 8 + i) * 512 + 256 + col] = (__bf16)v;
        mx = fmaxf(mx, v);
      }
      fgtmp[(wv * 2 + h) * 256 + col] = mx;
    }
  }
  __syncthreads();
  for (int c = tid; c < 256; c += 256) {
    float mx = -3.4e38f;
#pragma unroll
    for (int j = 0; j < 16; ++j) mx = fmaxf(mx, fgtmp[j * 256 + c]);
    fg[c] = mx;
  }
  __syncthreads();
  for (int e = tid; e < 128 * 256; e += 256) {  // broadcast fg into cols 0..255
    int r = e >> 8, c = e & 255;
    zb[(size_t)r * 512 + c] = (__bf16)fg[c];
  }
  __syncthreads();

  // ---- conv3: M=128 N=512 K=512 (WMMA, 2-wide N blocking) ----
  for (int nt = 0; nt < 32; nt += 2) {
    v8f acc0 = {0.f, 0.f, 0.f, 0.f, 0.f, 0.f, 0.f, 0.f};
    v8f acc1 = acc0;
#pragma unroll
    for (int ks = 0; ks < 16; ++ks) {
      v16bf a = frag_a_lds(zb, mr, ks * 32, 512);
      acc0 = wmma_bf16(a, frag_b_glb(w3bf, nt * 16, ks * 32, 512), acc0);
      acc1 = wmma_bf16(a, frag_b_glb(w3bf, (nt + 1) * 16, ks * 32, 512), acc1);
    }
#pragma unroll
    for (int t = 0; t < 2; ++t) {
      v8f acc = t ? acc1 : acc0;
      int col = (nt + t) * 16 + m;
      float bias = b3[col];
      if (WITH_OUT) {
        float sc = sc2l[col], sh = sh2l[col];
#pragma unroll
        for (int i = 0; i < 8; ++i) {
          float v = fmaxf((acc[i] + bias) * sc + sh, 0.f);
          a3[(size_t)(mr + h * 8 + i) * 512 + col] = (__bf16)v;
        }
      } else {
        float s = 0.f, ss = 0.f;
#pragma unroll
        for (int i = 0; i < 8; ++i) {
          float v = acc[i] + bias;
          s += v; ss += v * v;
        }
        statS[(wv * 2 + h) * 512 + col] = s;
        statSS[(wv * 2 + h) * 512 + col] = ss;
      }
    }
  }
  __syncthreads();

  if (!WITH_OUT) {  // deterministic per-group BN2 partials
    for (int c = tid; c < 512; c += 256) {
      float s = 0.f, ss = 0.f;
#pragma unroll
      for (int j = 0; j < 16; ++j) { s += statS[j * 512 + c]; ss += statSS[j * 512 + c]; }
      bn2part[(size_t)g * 1024 + c] = s;
      bn2part[(size_t)g * 1024 + 512 + c] = ss;
    }
    return;
  }

  // ---- conv4: M=128 N=128 K=512 (WMMA) + max over k -> feat ----
  float* ftmp = scr;  // [16][128]
  for (int nt = 0; nt < 8; nt += 2) {
    v8f acc0 = {0.f, 0.f, 0.f, 0.f, 0.f, 0.f, 0.f, 0.f};
    v8f acc1 = acc0;
#pragma unroll
    for (int ks = 0; ks < 16; ++ks) {
      v16bf a = frag_a_lds(a3, mr, ks * 32, 512);
      acc0 = wmma_bf16(a, frag_b_glb(w4bf, nt * 16, ks * 32, 512), acc0);
      acc1 = wmma_bf16(a, frag_b_glb(w4bf, (nt + 1) * 16, ks * 32, 512), acc1);
    }
#pragma unroll
    for (int t = 0; t < 2; ++t) {
      v8f acc = t ? acc1 : acc0;
      int col = (nt + t) * 16 + m;
      float mx = -3.4e38f;
#pragma unroll
      for (int i = 0; i < 8; ++i) mx = fmaxf(mx, acc[i]);
      ftmp[(wv * 2 + h) * 128 + col] = mx;
    }
  }
  __syncthreads();
  for (int c = tid; c < 128; c += 256) {
    float mx = -3.4e38f;
#pragma unroll
    for (int j = 0; j < 16; ++j) mx = fmaxf(mx, ftmp[j * 128 + c]);
    int bb = g >> 11, nn = g & (NPTS - 1);
    outfeat[((size_t)bb * 128 + c) * NPTS + nn] = mx + b4[c];
  }
}

__global__ __launch_bounds__(256) void reduce_bn2(const float* __restrict__ part,
                                                  float* __restrict__ acc) {
  int t = blockIdx.x * 256 + threadIdx.x;  // 0..1023
  float s = 0.f;
  for (int g = 0; g < NGRP; ++g) s += part[(size_t)g * 1024 + t];
  acc[t] = s;
}

__global__ void bn2_finalize(const float* __restrict__ acc,
                             const float* __restrict__ g2,
                             const float* __restrict__ be2,
                             float* __restrict__ sc2, float* __restrict__ sh2) {
  int c = threadIdx.x;
  if (c >= 512) return;
  float mean = acc[c] / CNT_BN;
  float var = acc[512 + c] / CNT_BN - mean * mean;
  float sc = g2[c] * rsqrtf(var + 1e-5f);
  sc2[c] = sc;
  sh2[c] = be2[c] - sc * mean;
}

// ---------------------------------------------------------------------------
extern "C" void kernel_launch(void* const* d_in, const int* in_sizes, int n_in,
                              void* d_out, int out_size, void* d_ws, size_t ws_size,
                              hipStream_t stream) {
  const float* pos = (const float*)d_in[0];
  const float* w1  = (const float*)d_in[1];
  const float* b1  = (const float*)d_in[2];
  const float* g1  = (const float*)d_in[3];
  const float* be1 = (const float*)d_in[4];
  const float* w2  = (const float*)d_in[5];
  const float* b2  = (const float*)d_in[6];
  const float* w3  = (const float*)d_in[7];
  const float* b3  = (const float*)d_in[8];
  const float* g2  = (const float*)d_in[9];
  const float* be2 = (const float*)d_in[10];
  const float* w4  = (const float*)d_in[11];
  const float* b4  = (const float*)d_in[12];

  char* ws = (char*)d_ws;
  float*  gp    = (float*)(ws + WS_GP);
  __bf16* w2bf  = (__bf16*)(ws + WS_W2);
  __bf16* w3bf  = (__bf16*)(ws + WS_W3);
  __bf16* w4bf  = (__bf16*)(ws + WS_W4);
  float*  fws   = (float*)(ws + WS_F);
  float*  sc1   = fws + FI_SC1;
  float*  sh1   = fws + FI_SH1;
  float*  sc2   = fws + FI_SC2;
  float*  sh2   = fws + FI_SH2;
  float*  st9   = fws + FI_STAT9;
  float*  bn2a  = fws + FI_BN2ACC;
  float*  bn1p  = fws + FI_BN1PART;
  float*  bn2p  = fws + FI_BN2PART;

  float* feat   = (float*)d_out;                      // (2,128,2048)
  int*   outidx = (int*)d_out + NBATCH * 128 * NPTS;  // (2,2048,128)

  convert_w<<<dim3(1408), dim3(256), 0, stream>>>(w2, w3, w4, w2bf, w3bf, w4bf);
  knn_kernel<<<dim3(NGRP), dim3(256), 0, stream>>>(pos, gp, outidx, bn1p);
  reduce_bn1<<<dim3(1), dim3(32), 0, stream>>>(bn1p, st9);
  bn1_finalize<<<dim3(1), dim3(128), 0, stream>>>(st9, w1, b1, g1, be1, sc1, sh1);
  stage_kernel<0><<<dim3(NGRP), dim3(256), SMEM_B, stream>>>(
      gp, w1, w2bf, w3bf, w4bf, sc1, sh1, b2, b3, sc2, sh2, b4, bn2p, feat);
  reduce_bn2<<<dim3(4), dim3(256), 0, stream>>>(bn2p, bn2a);
  bn2_finalize<<<dim3(1), dim3(512), 0, stream>>>(bn2a, g2, be2, sc2, sh2);
  stage_kernel<1><<<dim3(NGRP), dim3(256), SMEM_C, stream>>>(
      gp, w1, w2bf, w3bf, w4bf, sc1, sh1, b2, b3, sc2, sh2, b4, bn2p, feat);
}
